// QwenMoeWrapperSkip_32461362823834
// MI455X (gfx1250) — matrix operand order repaired and checked by reading, not compiled, not run on hardware
//
#include <hip/hip_runtime.h>
#include <hip/hip_bf16.h>

// ---------------------------------------------------------------------------
// QwenMoE (modality-skip) for gfx1250: bf16x3-split WMMA GEMMs, fused routing.
// Round 2: fragment-major LDS layouts -> vector ds_load_b128 fragment loads.
// ---------------------------------------------------------------------------

typedef __attribute__((ext_vector_type(16))) __bf16 v16bf;
typedef __attribute__((ext_vector_type(8)))  __bf16 v8bf;
typedef __attribute__((ext_vector_type(8)))  float  v8f;

#define BS_TOK 2048
#define HDIM   1024
#define NEXP   8
#define FDIM   512
#define TWOF   1024
#define KDN    4096   // NEXP * FDIM

// workspace layout (bytes)
#define OFF_VC   ((size_t)0)                                   // int[8]  valid counts
#define OFF_MC   ((size_t)64)                                  // int[8]  mismatch counts
#define OFF_SEL  ((size_t)256)                                 // int[BS*2]
#define OFF_RW   (OFF_SEL + (size_t)BS_TOK * 2 * sizeof(int))  // float[BS*2]
#define OFF_DW   (OFF_RW  + (size_t)BS_TOK * 2 * sizeof(float))// float[BS*NEXP]
#define OFF_A2HI (OFF_DW  + (size_t)BS_TOK * NEXP * sizeof(float)) // bf16[BS*KDN]
#define OFF_A2LO (OFF_A2HI + (size_t)BS_TOK * KDN * 2)             // bf16[BS*KDN]

__device__ __forceinline__ void split_bf16(float x, __bf16 &hi, __bf16 &lo) {
    hi = (__bf16)x;
    lo = (__bf16)(x - (float)hi);
}

// A fragment: 16x32 bf16 (M x K). LDS is row-major [rows][32].
// Lane layout: row = lane&15, K chunks at koff and koff+8,
// koff = 16*(lane>=16). Both halves are 16B-aligned contiguous chunks.
__device__ __forceinline__ v16bf load_a_frag(const __bf16* As, int lane, int mrow) {
    const int row  = mrow + (lane & 15);
    const int koff = (lane >> 4) * 16;
    const __bf16* p = As + row * 32;
    const v8bf a0 = *(const v8bf*)(p + koff);
    const v8bf a1 = *(const v8bf*)(p + 8 + koff);
    return __builtin_shufflevector(a0, a1, 0, 1, 2, 3, 4, 5, 6, 7,
                                           8, 9, 10, 11, 12, 13, 14, 15);
}

// B fragment: 32x16 bf16 (K x N). LDS is FRAGMENT-major: Bs[col][k], k=0..31.
// Lane layout: col = lane&15 (+col0), kb = 16*(lane>=16) -> 32 contiguous bytes.
__device__ __forceinline__ v16bf load_b_frag(const __bf16* Bs, int lane, int col0) {
    const int col = col0 + (lane & 15);
    const int kb  = (lane >> 4) * 16;
    return *(const v16bf*)(Bs + col * 32 + kb);
}

#define WMMA_BF16(acc, a, b) \
    (acc) = __builtin_amdgcn_wmma_f32_16x16x32_bf16(false, (a), false, (b), (short)0, (acc), false, false)

// 3-term bf16 split product: A*B ~= Ahi*Bhi + Ahi*Blo + Alo*Bhi
#define WMMA3(acc, ahi, alo, bhi, blo) \
    do { WMMA_BF16(acc, ahi, bhi); WMMA_BF16(acc, ahi, blo); WMMA_BF16(acc, alo, bhi); } while (0)

// ---------------------------------------------------------------------------
// K0: zero atomic counters
// ---------------------------------------------------------------------------
__global__ void moe_init_kernel(int* __restrict__ vc, int* __restrict__ mc) {
    const int t = threadIdx.x;
    if (t < NEXP) { vc[t] = 0; mc[t] = 0; }
}

// ---------------------------------------------------------------------------
// K1: router logits + top-2 + softmax-of-2 + modality counting
// ---------------------------------------------------------------------------
__global__ __launch_bounds__(256) void moe_router_kernel(
    const float* __restrict__ x, const float* __restrict__ gw,
    const int* __restrict__ tok_mod, const int* __restrict__ exp_mod,
    int* __restrict__ vc, int* __restrict__ mc,
    int* __restrict__ sel, float* __restrict__ rw)
{
    __shared__ float gws[NEXP * HDIM];  // 32 KB
    const int tid = threadIdx.x;
    for (int i = tid; i < NEXP * HDIM; i += 256) gws[i] = gw[i];
    __syncthreads();

    const int t = blockIdx.x * 256 + tid;
    const float* xr = x + (size_t)t * HDIM;

    float acc[NEXP];
#pragma unroll
    for (int e = 0; e < NEXP; ++e) acc[e] = 0.0f;
    for (int h = 0; h < HDIM; ++h) {
        const float xv = xr[h];
#pragma unroll
        for (int e = 0; e < NEXP; ++e) acc[e] += xv * gws[e * HDIM + h];
    }

    // top-2 (first index wins ties, matching jax.lax.top_k)
    int b0 = 0; float l0 = acc[0];
#pragma unroll
    for (int e = 1; e < NEXP; ++e) if (acc[e] > l0) { l0 = acc[e]; b0 = e; }
    int b1 = (b0 == 0) ? 1 : 0; float l1 = acc[b1];
#pragma unroll
    for (int e = 0; e < NEXP; ++e) if (e != b0 && acc[e] > l1) { l1 = acc[e]; b1 = e; }

    // softmax denominator cancels in top-k renorm -> 2-way softmax of logits
    const float eo = __expf(l1 - l0);
    const float w0 = 1.0f / (1.0f + eo);
    const float w1 = eo / (1.0f + eo);

    sel[t * 2 + 0] = b0;  sel[t * 2 + 1] = b1;
    rw[t * 2 + 0]  = w0;  rw[t * 2 + 1]  = w1;

    const int m = tok_mod[t];
    const int s2[2] = {b0, b1};
#pragma unroll
    for (int k = 0; k < 2; ++k) {
        const int em = exp_mod[s2[k]];
        if (m != 0 && em != 0) {
            atomicAdd(&vc[s2[k]], 1);
            if (m * em == -1) atomicAdd(&mc[s2[k]], 1);
        }
    }
}

// ---------------------------------------------------------------------------
// K2: expert skip decision + final renorm + dense routing weights [bs, E]
// ---------------------------------------------------------------------------
__global__ __launch_bounds__(256) void moe_finalize_kernel(
    const int* __restrict__ vc, const int* __restrict__ mc,
    const int* __restrict__ exp_mod,
    const int* __restrict__ sel, const float* __restrict__ rw,
    float* __restrict__ dw)
{
    __shared__ int skp[NEXP];
    const int tid = threadIdx.x;
    if (tid < NEXP)
        skp[tid] = (vc[tid] > 0) && (mc[tid] == vc[tid]) && (exp_mod[tid] != 0);
    __syncthreads();

    for (int t = tid; t < BS_TOK; t += 256) {
        const int s0 = sel[t * 2 + 0], s1 = sel[t * 2 + 1];
        float w0 = skp[s0] ? 0.0f : rw[t * 2 + 0];
        float w1 = skp[s1] ? 0.0f : rw[t * 2 + 1];
        const float s = w0 + w1;
        if (s > 0.0f) { const float d = fmaxf(s, 1e-9f); w0 /= d; w1 /= d; }
#pragma unroll
        for (int e = 0; e < NEXP; ++e)
            dw[t * NEXP + e] = (e == s0) ? w0 : ((e == s1) ? w1 : 0.0f);
    }
}

// ---------------------------------------------------------------------------
// K3: gate/up GEMM + SiLU + routing-weight scale -> A2 = w * silu(g)*u  (bf16 hi/lo)
// grid: (32 m-tiles, 8 f-tiles, 8 experts), 256 threads (8 wave32)
// per workgroup: 64 tokens x 64 f-cols, K-loop over HDIM in steps of 32
// ---------------------------------------------------------------------------
__global__ __launch_bounds__(256) void moe_gu_kernel(
    const float* __restrict__ x, const float* __restrict__ gup,
    const float* __restrict__ dw,
    __bf16* __restrict__ a2hi, __bf16* __restrict__ a2lo)
{
    __shared__ __attribute__((aligned(16))) __bf16 As_hi[64 * 32], As_lo[64 * 32]; // 8 KB
    // B tiles stored FRAGMENT-major: Bs[col][k]  (col 0..127, k 0..31)
    __shared__ __attribute__((aligned(16))) __bf16 Bs_hi[128 * 32], Bs_lo[128 * 32]; // 16 KB
    __shared__ float wrow[64];

    const int m0   = blockIdx.x * 64;
    const int f0   = blockIdx.y * 64;
    const int e    = blockIdx.z;
    const int tid  = threadIdx.x;
    const int lane = tid & 31;
    const int w    = tid >> 5;
    const int mrow = (w & 3) * 16;     // row block within 64
    const int ch   = w >> 2;           // column half: 0 -> cols 0..31, 1 -> 32..63
    const size_t eoff = (size_t)e * HDIM * TWOF;

    if (tid < 64) wrow[tid] = dw[(size_t)(m0 + tid) * NEXP + e];

    const v8f vzero = {0.f, 0.f, 0.f, 0.f, 0.f, 0.f, 0.f, 0.f};
    v8f acc[4];   // [0..1]=gate tiles, [2..3]=up tiles
#pragma unroll
    for (int j = 0; j < 4; ++j) acc[j] = vzero;

    for (int k0 = 0; k0 < HDIM; k0 += 32) {
        __syncthreads();
        // stage A: x[m0..m0+63][k0..k0+31], split to bf16 hi/lo (row-major)
#pragma unroll
        for (int i = 0; i < 8; ++i) {
            const int lin = i * 256 + tid;
            const int r = lin >> 5, c = lin & 31;
            const float v = x[(size_t)(m0 + r) * HDIM + k0 + c];
            __bf16 hi, lo; split_bf16(v, hi, lo);
            As_hi[r * 32 + c] = hi; As_lo[r * 32 + c] = lo;
        }
        // stage B transposed: gate cols f0..f0+63 (c<64) and up cols FDIM+f0.. (c>=64)
        // global read coalesced over c; LDS store at Bs[c][kk]
#pragma unroll
        for (int i = 0; i < 16; ++i) {
            const int lin = i * 256 + tid;
            const int kk = lin >> 7, c = lin & 127;
            const int col = (c < 64) ? (f0 + c) : (FDIM + f0 + (c - 64));
            const float v = gup[eoff + (size_t)(k0 + kk) * TWOF + col];
            __bf16 hi, lo; split_bf16(v, hi, lo);
            Bs_hi[c * 32 + kk] = hi; Bs_lo[c * 32 + kk] = lo;
        }
        if (k0 + 32 < HDIM) {
            __builtin_prefetch(&x[(size_t)(m0 + (tid >> 5) * 8) * HDIM + k0 + 32], 0, 1);
            __builtin_prefetch(&gup[eoff + (size_t)(k0 + 32 + (tid & 31)) * TWOF + f0], 0, 1);
        }
        __syncthreads();

        const v16bf ahi = load_a_frag(As_hi, lane, mrow);
        const v16bf alo = load_a_frag(As_lo, lane, mrow);
#pragma unroll
        for (int j = 0; j < 4; ++j) {
            // j=0,1 -> gate cols ch*32 + {0,16}; j=2,3 -> up cols 64 + ch*32 + {0,16}
            const int col0 = (j >> 1) * 64 + ch * 32 + (j & 1) * 16;
            const v16bf bhi = load_b_frag(Bs_hi, lane, col0);
            const v16bf blo = load_b_frag(Bs_lo, lane, col0);
            WMMA3(acc[j], ahi, alo, bhi, blo);
        }
    }

    // epilogue: act = silu(g)*u, scale by routing weight, store bf16 hi/lo
    const int colb = lane & 15;
    const int voff = (lane >> 4) * 8;
#pragma unroll
    for (int j = 0; j < 2; ++j) {
        const int fcol = f0 + ch * 32 + j * 16 + colb;       // [0, FDIM)
#pragma unroll
        for (int v = 0; v < 8; ++v) {
            const int r   = mrow + voff + v;                 // row within 64
            const int row = m0 + r;                          // token index
            const float g = acc[j][v];
            const float u = acc[2 + j][v];
            const float s = g / (1.0f + __expf(-g));         // silu
            const float val = wrow[r] * s * u;
            __bf16 hi, lo; split_bf16(val, hi, lo);
            const size_t idx = (size_t)row * KDN + e * FDIM + fcol;
            a2hi[idx] = hi; a2lo[idx] = lo;
        }
    }
}

// ---------------------------------------------------------------------------
// K4: out = A2 @ down_proj.reshape(4096, 1024)   (expert sum folded into K)
// grid: (32 m-tiles, 16 h-tiles), 256 threads, K-loop over 4096 in steps of 32
// ---------------------------------------------------------------------------
__global__ __launch_bounds__(256) void moe_down_kernel(
    const __bf16* __restrict__ a2hi, const __bf16* __restrict__ a2lo,
    const float* __restrict__ down, float* __restrict__ out)
{
    __shared__ __attribute__((aligned(16))) __bf16 Ah[64 * 32], Al[64 * 32]; // 8 KB
    // B tile FRAGMENT-major: B[col][k]  (col 0..63, k 0..31)
    __shared__ __attribute__((aligned(16))) __bf16 Bh[64 * 32], Bl[64 * 32]; // 8 KB

    const int m0   = blockIdx.x * 64;
    const int h0   = blockIdx.y * 64;
    const int tid  = threadIdx.x;
    const int lane = tid & 31;
    const int w    = tid >> 5;
    const int mrow = (w & 3) * 16;
    const int ch   = w >> 2;

    const v8f vzero = {0.f, 0.f, 0.f, 0.f, 0.f, 0.f, 0.f, 0.f};
    v8f acc[2] = {vzero, vzero};

    for (int k0 = 0; k0 < KDN; k0 += 32) {
        __syncthreads();
        // stage A (already bf16 hi/lo in workspace), row-major
#pragma unroll
        for (int i = 0; i < 8; ++i) {
            const int lin = i * 256 + tid;
            const int r = lin >> 5, c = lin & 31;
            const size_t idx = (size_t)(m0 + r) * KDN + k0 + c;
            Ah[r * 32 + c] = a2hi[idx];
            Al[r * 32 + c] = a2lo[idx];
        }
        // stage B transposed: down[k0+kk][h0+c] -> B[c][kk]
#pragma unroll
        for (int i = 0; i < 8; ++i) {
            const int lin = i * 256 + tid;
            const int kk = lin >> 6, c = lin & 63;
            const float v = down[(size_t)(k0 + kk) * HDIM + h0 + c];
            __bf16 hi, lo; split_bf16(v, hi, lo);
            Bh[c * 32 + kk] = hi; Bl[c * 32 + kk] = lo;
        }
        if (k0 + 32 < KDN) {
            __builtin_prefetch(&a2hi[(size_t)(m0 + (tid >> 5) * 8) * KDN + k0 + 32], 0, 1);
            __builtin_prefetch(&down[(size_t)(k0 + 32 + (tid & 31)) * HDIM + h0], 0, 1);
        }
        __syncthreads();

        const v16bf ahi = load_a_frag(Ah, lane, mrow);
        const v16bf alo = load_a_frag(Al, lane, mrow);
#pragma unroll
        for (int j = 0; j < 2; ++j) {
            const int col0 = ch * 32 + j * 16;
            const v16bf bhi = load_b_frag(Bh, lane, col0);
            const v16bf blo = load_b_frag(Bl, lane, col0);
            WMMA3(acc[j], ahi, alo, bhi, blo);
        }
    }

    const int colb = lane & 15;
    const int voff = (lane >> 4) * 8;
#pragma unroll
    for (int j = 0; j < 2; ++j) {
        const int hcol = h0 + ch * 32 + j * 16 + colb;
#pragma unroll
        for (int v = 0; v < 8; ++v) {
            const int row = m0 + mrow + voff + v;
            out[(size_t)row * HDIM + hcol] = acc[j][v];
        }
    }
}

// ---------------------------------------------------------------------------
extern "C" void kernel_launch(void* const* d_in, const int* in_sizes, int n_in,
                              void* d_out, int out_size, void* d_ws, size_t ws_size,
                              hipStream_t stream) {
    const float* x    = (const float*)d_in[0];   // [2,1024,1024] -> [2048,1024]
    const float* gw   = (const float*)d_in[1];   // [8,1024]
    const float* gup  = (const float*)d_in[2];   // [8,1024,1024]
    const float* dpj  = (const float*)d_in[3];   // [8,512,1024] == [4096,1024]
    const int*   tmod = (const int*)d_in[4];     // [2048]
    const int*   emod = (const int*)d_in[5];     // [8]
    float* out = (float*)d_out;

    char* ws = (char*)d_ws;
    int*    vc   = (int*)(ws + OFF_VC);
    int*    mc   = (int*)(ws + OFF_MC);
    int*    sel  = (int*)(ws + OFF_SEL);
    float*  rw   = (float*)(ws + OFF_RW);
    float*  dw   = (float*)(ws + OFF_DW);
    __bf16* a2hi = (__bf16*)(ws + OFF_A2HI);
    __bf16* a2lo = (__bf16*)(ws + OFF_A2LO);

    moe_init_kernel<<<1, 32, 0, stream>>>(vc, mc);
    moe_router_kernel<<<BS_TOK / 256, 256, 0, stream>>>(x, gw, tmod, emod, vc, mc, sel, rw);
    moe_finalize_kernel<<<1, 256, 0, stream>>>(vc, mc, emod, sel, rw, dw);
    moe_gu_kernel<<<dim3(BS_TOK / 64, FDIM / 64, NEXP), 256, 0, stream>>>(x, gup, dw, a2hi, a2lo);
    moe_down_kernel<<<dim3(BS_TOK / 64, HDIM / 64), 256, 0, stream>>>(a2hi, a2lo, dpj, out);
}
